// S4TTS_6957847020308
// MI455X (gfx1250) — matrix operand order repaired and checked by reading, not compiled, not run on hardware
//
#include <hip/hip_runtime.h>
#include <math.h>

typedef unsigned short u16;
typedef unsigned int   u32;
typedef __bf16 bf16;
typedef bf16  v16bf __attribute__((ext_vector_type(16)));
typedef float v8f   __attribute__((ext_vector_type(8)));

static constexpr int HD = 384;   // d_model
static constexpr int NS = 32;    // complex state pairs
static constexpr int BB_ = 8;    // batch

__device__ __forceinline__ u16 f2bf(float f) {
  u32 u = __float_as_uint(f);
  u32 r = u + 0x7FFFu + ((u >> 16) & 1u);   // round-to-nearest-even
  return (u16)(r >> 16);
}

__device__ __forceinline__ float gelu_f(float x) {
  return 0.5f * x * (1.0f + erff(x * 0.70710678118654752f));
}

// ---------------- small prep kernels ----------------

__global__ void k_embed(const int* __restrict__ x, const float* __restrict__ emb,
                        u16* __restrict__ out, int total) {
  int t = blockIdx.x * blockDim.x + threadIdx.x;
  if (t >= total) return;
  int c = t % HD, r = t / HD;
  out[t] = f2bf(emb[(size_t)x[r] * HD + c]);
}

// W (K,N) f32 -> Wt (N,K) bf16
__global__ void k_cvt_t(const float* __restrict__ W, u16* __restrict__ Wt, int K, int N) {
  int t = blockIdx.x * blockDim.x + threadIdx.x;
  if (t >= K * N) return;
  int n = t / K, k = t % K;
  Wt[t] = f2bf(W[(size_t)k * N + n]);
}

// ConvTranspose1d weight (C=384, O=384, K=4) -> Wt (1536, 384) bf16, col n = k*384+o
__global__ void k_cvt_up(const float* __restrict__ W, u16* __restrict__ Wt) {
  int t = blockIdx.x * blockDim.x + threadIdx.x;
  if (t >= 1536 * HD) return;
  int n = t / HD, c = t % HD;
  int o = n % HD, k = n / HD;
  Wt[t] = f2bf(W[((size_t)c * HD + o) * 4 + k]);
}

// Discretize: lambda=exp(dt*A), C' = C*(lambda-1)/A, lamM = lambda^chunkLen
__global__ void k_prep(const float* __restrict__ log_dt, const float* __restrict__ Arl,
                       const float* __restrict__ Aim, const float* __restrict__ Cre,
                       const float* __restrict__ Cim,
                       float* Lre, float* Lim, float* Pr, float* Pi, float* Mre, float* Mim,
                       int layerOff, float chunkLen) {
  int t = blockIdx.x * blockDim.x + threadIdx.x;
  if (t >= 6 * HD * NS) return;
  int n = t % NS, h = (t / NS) % HD, l = t / (NS * HD);
  float dt = expf(log_dt[l * HD + h]);
  float ar = -expf(Arl[t]);
  float ai = Aim[t];
  float dr = dt * ar, di = dt * ai;
  float er = expf(dr);
  float lr = er * cosf(di), li = er * sinf(di);
  float em = expf(chunkLen * dr);
  float am = chunkLen * di;
  float mr = em * cosf(am), mi = em * sinf(am);
  float wr = lr - 1.0f, wi = li;
  float den = ar * ar + ai * ai;
  float qr = (wr * ar + wi * ai) / den, qi = (wi * ar - wr * ai) / den;
  float cr = Cre[t], ci = Cim[t];
  int g = ((layerOff + l) * HD + h) * NS + n;
  Lre[g] = lr; Lim[g] = li;
  Pr[g] = cr * qr - ci * qi;
  Pi[g] = cr * qi + ci * qr;
  Mre[g] = mr; Mim[g] = mi;
}

// ---------------- chunked SSM scan ----------------

// Pass 1: chunk-local end states with zero init
__global__ void k_scan_pass1(const float* __restrict__ u,
                             const float* __restrict__ Lre, const float* __restrict__ Lim,
                             float* __restrict__ s_end,
                             int layer, int Lseq, int nchunks, int chunkLen) {
  int t = blockIdx.x * blockDim.x + threadIdx.x;
  if (t >= BB_ * HD * nchunks) return;
  int h = t % HD, b = (t / HD) % BB_, c = t / (HD * BB_);
  float lr[NS], li[NS], sr[NS], si[NS];
  int pb = (layer * HD + h) * NS;
  #pragma unroll
  for (int n = 0; n < NS; ++n) { lr[n] = Lre[pb + n]; li[n] = Lim[pb + n]; sr[n] = 0.f; si[n] = 0.f; }
  const float* up = u + (size_t)b * Lseq * HD + h;
  int l0 = c * chunkLen, l1 = l0 + chunkLen;
  for (int l = l0; l < l1; ++l) {
    float uv = up[(size_t)l * HD];
    #pragma unroll
    for (int n = 0; n < NS; ++n) {
      float tr = lr[n] * sr[n] - li[n] * si[n] + uv;
      si[n] = lr[n] * si[n] + li[n] * sr[n];
      sr[n] = tr;
    }
  }
  size_t base = (((size_t)(b * HD + h)) * nchunks + c) * 64;
  #pragma unroll
  for (int n = 0; n < NS; ++n) { s_end[base + n] = sr[n]; s_end[base + 32 + n] = si[n]; }
}

// Fixup: propagate incoming states across chunk boundaries
__global__ void k_scan_fix(const float* __restrict__ Mre, const float* __restrict__ Mim,
                           const float* __restrict__ s_end, float* __restrict__ S_in,
                           int layer, int nchunks) {
  int t = blockIdx.x * blockDim.x + threadIdx.x;
  if (t >= BB_ * HD) return;
  int h = t % HD, b = t / HD;
  float mr[NS], mi[NS], Sr[NS], Si[NS];
  int pb = (layer * HD + h) * NS;
  #pragma unroll
  for (int n = 0; n < NS; ++n) { mr[n] = Mre[pb + n]; mi[n] = Mim[pb + n]; Sr[n] = 0.f; Si[n] = 0.f; }
  size_t rowb = ((size_t)(b * HD + h)) * nchunks;
  for (int c = 1; c < nchunks; ++c) {
    size_t pp = (rowb + c - 1) * 64, pc = (rowb + c) * 64;
    #pragma unroll
    for (int n = 0; n < NS; ++n) {
      float ser = s_end[pp + n], sei = s_end[pp + 32 + n];
      float tr = mr[n] * Sr[n] - mi[n] * Si[n] + ser;
      Si[n] = mr[n] * Si[n] + mi[n] * Sr[n] + sei;
      Sr[n] = tr;
      S_in[pc + n] = Sr[n]; S_in[pc + 32 + n] = Si[n];
    }
  }
}

// Pass 2: full scan per chunk with correct incoming state; fused D-skip + GELU + bf16
__global__ void k_scan_pass2(const float* __restrict__ u,
                             const float* __restrict__ Lre, const float* __restrict__ Lim,
                             const float* __restrict__ Pr, const float* __restrict__ Pi,
                             const float* __restrict__ Dl,
                             const float* __restrict__ S_in,
                             u16* __restrict__ ybf,
                             int layer, int Lseq, int nchunks, int chunkLen) {
  int t = blockIdx.x * blockDim.x + threadIdx.x;
  if (t >= BB_ * HD * nchunks) return;
  int h = t % HD, b = (t / HD) % BB_, c = t / (HD * BB_);
  float lr[NS], li[NS], cr[NS], ci[NS], sr[NS], si[NS];
  int pb = (layer * HD + h) * NS;
  #pragma unroll
  for (int n = 0; n < NS; ++n) {
    lr[n] = Lre[pb + n]; li[n] = Lim[pb + n];
    cr[n] = Pr[pb + n];  ci[n] = Pi[pb + n];
  }
  if (c == 0) {
    #pragma unroll
    for (int n = 0; n < NS; ++n) { sr[n] = 0.f; si[n] = 0.f; }
  } else {
    size_t base = (((size_t)(b * HD + h)) * nchunks + c) * 64;
    #pragma unroll
    for (int n = 0; n < NS; ++n) { sr[n] = S_in[base + n]; si[n] = S_in[base + 32 + n]; }
  }
  float dv = Dl[h];
  const float* up = u + (size_t)b * Lseq * HD + h;
  u16* yp = ybf + (size_t)b * Lseq * HD + h;
  int l0 = c * chunkLen, l1 = l0 + chunkLen;
  for (int l = l0; l < l1; ++l) {
    float uv = up[(size_t)l * HD];
    float acc = 0.f;
    #pragma unroll
    for (int n = 0; n < NS; ++n) {
      float tr = lr[n] * sr[n] - li[n] * si[n] + uv;
      si[n] = lr[n] * si[n] + li[n] * sr[n];
      sr[n] = tr;
      acc += cr[n] * sr[n] - ci[n] * si[n];
    }
    float y = 2.f * acc + dv * uv;
    yp[(size_t)l * HD] = f2bf(gelu_f(y));
  }
}

// ---------------- WMMA GEMM (bf16 in, f32 acc), fused epilogues ----------------
// EPI: 0=bias  1=bias+GELU  2=GLU (WT has 2N rows, dual accumulator)  3=bias[col%384]
// NT: N-subtiles per wave (16xNT*16 output tile) -> A fragment reused NT (2*NT for GLU) times

union AFrag { u32 u[8]; v16bf v; };

template<int EPI, int WBF, int NT>
__global__ __launch_bounds__(256) void k_gemm(
    const u16* __restrict__ A, const u16* __restrict__ WT,
    const float* __restrict__ bias,
    float* __restrict__ outF, u16* __restrict__ outB,
    int M, int N, int K, int tilesNg) {
  int wid = blockIdx.x * (blockDim.x >> 5) + (threadIdx.x >> 5);
  int tiles = (M >> 4) * tilesNg;
  if (wid >= tiles) return;
  int lane = threadIdx.x & 31;
  int hl = lane >> 4, l15 = lane & 15;
  int tm = wid / tilesNg, tn = wid % tilesNg;
  const u16* Ar = A + (size_t)((tm << 4) + l15) * K;
  const u16* Brj[NT];
  #pragma unroll
  for (int j = 0; j < NT; ++j)
    Brj[j] = WT + (size_t)(((tn * NT + j) << 4) + l15) * K;
  v8f acc[NT], acc2[NT];
  #pragma unroll
  for (int j = 0; j < NT; ++j) {
    acc[j] = (v8f){0.f, 0.f, 0.f, 0.f, 0.f, 0.f, 0.f, 0.f};
    acc2[j] = acc[j];
  }
  for (int kb = 0; kb < K; kb += 32) {
    AFrag a;
    #pragma unroll
    for (int v = 0; v < 8; ++v) {
      int ka = kb + 2 * v + hl * 16;  // 16-bit A 16x32, contiguous k = 16h + s (matches B)
      a.u[v] = *(const u32*)(Ar + ka);
    }
    #pragma unroll
    for (int j = 0; j < NT; ++j) {
      AFrag bf;
      #pragma unroll
      for (int v = 0; v < 8; ++v) {
        int kc = kb + 2 * v + hl * 16;                                  // 16-bit B 32x16 layout
        bf.u[v] = *(const u32*)(Brj[j] + kc);
      }
      acc[j] = __builtin_amdgcn_wmma_f32_16x16x32_bf16(false, a.v, false, bf.v, (short)0,
                                                       acc[j], false, false);
      if (EPI == 2) {
        AFrag b2;
        #pragma unroll
        for (int v = 0; v < 8; ++v) {
          int kc = kb + 2 * v + hl * 16;
          b2.u[v] = *(const u32*)(Brj[j] + (size_t)N * K + kc);
        }
        acc2[j] = __builtin_amdgcn_wmma_f32_16x16x32_bf16(false, a.v, false, b2.v, (short)0,
                                                          acc2[j], false, false);
      }
    }
  }
  int row0 = (tm << 4) + hl * 8;
  #pragma unroll
  for (int j = 0; j < NT; ++j) {
    int col = ((tn * NT + j) << 4) + l15;
    #pragma unroll
    for (int r = 0; r < 8; ++r) {
      float val;
      if (EPI == 2) {
        float z1 = acc[j][r] + bias[col];
        float z2 = acc2[j][r] + bias[col + N];
        val = z1 * (1.0f / (1.0f + __expf(-z2)));
      } else if (EPI == 1) {
        val = gelu_f(acc[j][r] + bias[col]);
      } else if (EPI == 3) {
        val = acc[j][r] + bias[col % HD];
      } else {
        val = acc[j][r] + bias[col];
      }
      size_t o = (size_t)(row0 + r) * N + col;
      if (outF) outF[o] = val;
      if (WBF)  outB[o] = f2bf(val);
    }
  }
}

// ---------------- LayerNorm + GELU (one wave per 384-row) ----------------

__global__ void k_ln_gelu(const float* __restrict__ in, float* __restrict__ outF,
                          u16* __restrict__ outB,
                          const float* __restrict__ g, const float* __restrict__ bta, int rows) {
  int row = blockIdx.x * (blockDim.x >> 5) + (threadIdx.x >> 5);
  if (row >= rows) return;
  int lane = threadIdx.x & 31;
  const float* p = in + (size_t)row * HD;
  float x[12];
  float s = 0.f, s2 = 0.f;
  #pragma unroll
  for (int i = 0; i < 12; ++i) { x[i] = p[lane + i * 32]; s += x[i]; s2 += x[i] * x[i]; }
  #pragma unroll
  for (int off = 16; off >= 1; off >>= 1) {
    s  += __shfl_xor(s,  off, 32);
    s2 += __shfl_xor(s2, off, 32);
  }
  float m = s * (1.f / HD);
  float var = s2 * (1.f / HD) - m * m;
  float rs = rsqrtf(var + 1e-5f);
  #pragma unroll
  for (int i = 0; i < 12; ++i) {
    int cidx = lane + i * 32;
    float y = gelu_f((x[i] - m) * rs * g[cidx] + bta[cidx]);
    if (outF) outF[(size_t)row * HD + cidx] = y;
    if (outB) outB[(size_t)row * HD + cidx] = f2bf(y);
  }
}

// ---------------- orchestration ----------------

static void launch_gemm(int epi, const u16* A, const u16* WT, const float* bias,
                        float* oF, u16* oB, int M, int N, int K, hipStream_t s) {
  if ((N & 63) == 0) {
    int tilesNg = N >> 6;              // groups of 4 subtiles
    int tiles = (M >> 4) * tilesNg;
    int blocks = (tiles + 7) >> 3;
    if (epi == 1)      k_gemm<1, 0, 4><<<blocks, 256, 0, s>>>(A, WT, bias, oF, oB, M, N, K, tilesNg);
    else if (epi == 2) k_gemm<2, 1, 4><<<blocks, 256, 0, s>>>(A, WT, bias, oF, oB, M, N, K, tilesNg);
    else if (epi == 3) k_gemm<3, 0, 4><<<blocks, 256, 0, s>>>(A, WT, bias, oF, oB, M, N, K, tilesNg);
    else               k_gemm<0, 0, 4><<<blocks, 256, 0, s>>>(A, WT, bias, oF, oB, M, N, K, tilesNg);
  } else {
    int tilesNg = N >> 4;
    int tiles = (M >> 4) * tilesNg;
    int blocks = (tiles + 7) >> 3;
    k_gemm<0, 0, 1><<<blocks, 256, 0, s>>>(A, WT, bias, oF, oB, M, N, K, tilesNg);
  }
}

extern "C" void kernel_launch(void* const* d_in, const int* in_sizes, int n_in,
                              void* d_out, int out_size, void* d_ws, size_t ws_size,
                              hipStream_t stream) {
  (void)in_sizes; (void)n_in; (void)out_size; (void)ws_size;
  const int*   x      = (const int*)d_in[0];
  const float* embedW = (const float*)d_in[1];
  const float* pre_W  = (const float*)d_in[2];
  const float* pre_b  = (const float*)d_in[3];
  const float* e_logdt = (const float*)d_in[4];
  const float* e_Arl   = (const float*)d_in[5];
  const float* e_Aim   = (const float*)d_in[6];
  const float* e_Cre   = (const float*)d_in[7];
  const float* e_Cim   = (const float*)d_in[8];
  const float* e_D     = (const float*)d_in[9];
  const float* e_Wout  = (const float*)d_in[10];
  const float* e_bout  = (const float*)d_in[11];
  const float* up1_W = (const float*)d_in[12];
  const float* up1_b = (const float*)d_in[13];
  const float* ln1_g = (const float*)d_in[14];
  const float* ln1_b = (const float*)d_in[15];
  const float* up2_W = (const float*)d_in[16];
  const float* up2_b = (const float*)d_in[17];
  const float* ln2_g = (const float*)d_in[18];
  const float* ln2_b = (const float*)d_in[19];
  const float* d_logdt = (const float*)d_in[20];
  const float* d_Arl   = (const float*)d_in[21];
  const float* d_Aim   = (const float*)d_in[22];
  const float* d_Cre   = (const float*)d_in[23];
  const float* d_Cim   = (const float*)d_in[24];
  const float* d_D     = (const float*)d_in[25];
  const float* d_Wout  = (const float*)d_in[26];
  const float* d_bout  = (const float*)d_in[27];
  const float* out_W = (const float*)d_in[28];
  const float* out_b = (const float*)d_in[29];

  // workspace layout (~229 MB)
  char* ws = (char*)d_ws;
  float* G    = (float*)(ws);                      // 65536x384 f32     (100663296 B)
  u16*   BBuf = (u16*)(ws + 100663296UL);          // 65536x384 bf16    ( 50331648 B)
  u16*   BB2  = (u16*)(ws + 150994944UL);          // 65536x384 bf16    ( 50331648 B)
  u16*   WtPre = (u16*)(ws + 201326592UL);         // 384x384
  u16*   WtOut = (u16*)(ws + 201621504UL);         // 12 x 768x384
  u16*   WtUp1 = (u16*)(ws + 208699392UL);         // 1536x384
  u16*   WtUp2 = (u16*)(ws + 209879040UL);         // 1536x384
  u16*   WtO   = (u16*)(ws + 211058688UL);         // 80x384
  float* Lre = (float*)(ws + 211120128UL);         // 6 arrays of 12*384*32 f32
  float* Lim = Lre + 147456;
  float* Pr  = Lim + 147456;
  float* Pi  = Pr + 147456;
  float* Mre = Pi + 147456;
  float* Mim = Mre + 147456;
  float* Send = (float*)(ws + 214659072UL);        // 8*384*16*64 f32
  float* Sin  = (float*)(ws + 227241984UL);        // same size

  // --- weight conversion (f32 -> transposed bf16) ---
  k_cvt_t<<<(384 * 384 + 255) / 256, 256, 0, stream>>>(pre_W, WtPre, 384, 384);
  for (int l = 0; l < 6; ++l)
    k_cvt_t<<<(384 * 768 + 255) / 256, 256, 0, stream>>>(
        e_Wout + (size_t)l * 384 * 768, WtOut + (size_t)l * 768 * 384, 384, 768);
  for (int l = 0; l < 6; ++l)
    k_cvt_t<<<(384 * 768 + 255) / 256, 256, 0, stream>>>(
        d_Wout + (size_t)l * 384 * 768, WtOut + (size_t)(6 + l) * 768 * 384, 384, 768);
  k_cvt_up<<<(1536 * 384 + 255) / 256, 256, 0, stream>>>(up1_W, WtUp1);
  k_cvt_up<<<(1536 * 384 + 255) / 256, 256, 0, stream>>>(up2_W, WtUp2);
  k_cvt_t<<<(384 * 80 + 255) / 256, 256, 0, stream>>>(out_W, WtO, 384, 80);

  // --- SSM parameter discretization (12 layers) ---
  k_prep<<<(6 * HD * NS + 255) / 256, 256, 0, stream>>>(
      e_logdt, e_Arl, e_Aim, e_Cre, e_Cim, Lre, Lim, Pr, Pi, Mre, Mim, 0, 512.f);
  k_prep<<<(6 * HD * NS + 255) / 256, 256, 0, stream>>>(
      d_logdt, d_Arl, d_Aim, d_Cre, d_Cim, Lre, Lim, Pr, Pi, Mre, Mim, 6, 512.f);

  // --- embed + prenet ---
  k_embed<<<(4096 * HD + 255) / 256, 256, 0, stream>>>(x, embedW, BBuf, 4096 * HD);
  launch_gemm(1, BBuf, WtPre, pre_b, G, nullptr, 4096, 384, 384, stream);  // GELU epilogue

  // --- encoder: 6 S4 layers @ L=512 (single chunk) ---
  for (int l = 0; l < 6; ++l) {
    k_scan_pass2<<<12, 256, 0, stream>>>(G, Lre, Lim, Pr, Pi, e_D + l * HD, Sin, BBuf,
                                         l, 512, 1, 512);
    launch_gemm(2, BBuf, WtOut + (size_t)l * 768 * 384, e_bout + l * 768,
                G, BB2, 4096, 384, 384, stream);                           // GLU epilogue
  }

  // --- upsample 1 (4096 -> 16384 rows), LN+GELU ---
  launch_gemm(3, BB2, WtUp1, up1_b, G, nullptr, 4096, 1536, 384, stream);
  k_ln_gelu<<<16384 / 8, 256, 0, stream>>>(G, nullptr, BBuf, ln1_g, ln1_b, 16384);

  // --- upsample 2 (16384 -> 65536 rows), LN+GELU (in-place f32) ---
  launch_gemm(3, BBuf, WtUp2, up2_b, G, nullptr, 16384, 1536, 384, stream);
  k_ln_gelu<<<65536 / 8, 256, 0, stream>>>(G, G, nullptr, ln2_g, ln2_b, 65536);

  // --- decoder: 6 S4 layers @ L=8192, 16-way chunked scan ---
  for (int l = 0; l < 6; ++l) {
    k_scan_pass1<<<192, 256, 0, stream>>>(G, Lre, Lim, Send, 6 + l, 8192, 16, 512);
    k_scan_fix<<<12, 256, 0, stream>>>(Mre, Mim, Send, Sin, 6 + l, 16);
    k_scan_pass2<<<192, 256, 0, stream>>>(G, Lre, Lim, Pr, Pi, d_D + l * HD, Sin, BBuf,
                                          6 + l, 8192, 16, 512);
    launch_gemm(2, BBuf, WtOut + (size_t)(6 + l) * 768 * 384, d_bout + l * 768,
                G, BB2, 65536, 384, 384, stream);
  }

  // --- output projection to 80 mels ---
  launch_gemm(0, BB2, WtO, out_b, (float*)d_out, nullptr, 65536, 80, 384, stream);
}
